// CausalAttention_50629074485540
// MI455X (gfx1250) — matrix operand order repaired and checked by reading, not compiled, not run on hardware
//
#include <hip/hip_runtime.h>

// ---------------------------------------------------------------------------
// Causal attention with ALiBi for MI455X (gfx1250), wave32 + WMMA bf16 + TDM.
// B=2, T=2048, D=1024, H=16, hd=64.
// Pipeline:
//   1) cvt_bf16: x f32 -> bf16 ; cvt_w_t: W* f32 -> bf16 TRANSPOSED (N x K)
//   2) gemm_bf16<1/2> (x3): Q/K projections -> bf16 (B,H,T,hd),
//      V projection -> bf16 (B,H,hd,T)  (pre-transposed for flash)
//   3) flash_alibi: fused scores + ALiBi + causal + online softmax + P@V.
//      K/V tiles double-buffered in LDS via the Tensor Data Mover (wave 0
//      issues next step's tiles before compute; s_wait_tensorcnt at iteration
//      end -> DMA overlaps WMMA). All matmuls v_wmma_f32_16x16x32_bf16.
//   4) gemm_bf16<0>: attn_out @ Wo -> f32 d_out
// ---------------------------------------------------------------------------

typedef __attribute__((ext_vector_type(16))) __bf16 bf16x16;
typedef __attribute__((ext_vector_type(8)))  float  floatx8;
typedef __attribute__((ext_vector_type(4)))  unsigned int uint32x4;
typedef __attribute__((ext_vector_type(4)))  int          int32x4;
typedef __attribute__((ext_vector_type(8)))  int          int32x8;

#if defined(__has_builtin)
#if __has_builtin(__builtin_amdgcn_tensor_load_to_lds)
#define USE_TDM 1
#endif
#endif
#ifndef USE_TDM
#define USE_TDM 0
#endif
// therock-10.0 HIP headers (6-arg builtin) ship the TDM header; ROCm 7.2 = 5-arg
#if __has_include(<hip/amd_detail/amd_gfx1250_TDM.h>)
#define TDM_6ARG 1
#else
#define TDM_6ARG 0
#endif

union Frag {
  bf16x16      v;
  unsigned int u[8];
  __bf16       h[16];
  uint4        q[2];
};
union V8 {
  uint4  q;
  __bf16 h[8];
};

__device__ __forceinline__ floatx8 wmma_bf16(bf16x16 a, bf16x16 b, floatx8 c) {
  return __builtin_amdgcn_wmma_f32_16x16x32_bf16(
      /*neg_a=*/false, a, /*neg_b=*/false, b,
      /*c_mod=*/(short)0, c, /*reuse_a=*/false, /*reuse_b=*/false);
}

#define T_SEQ  2048
#define DMODEL 1024
#define NHEAD  16
#define HDIM   64
#define MROWS  4096  // B*T

#if USE_TDM
// 2D tile load via Tensor Data Mover, 2-byte elements, compacted into LDS.
// D# group layouts per CDNA5 ISA ch.8 (CDNA5 ISA).
__device__ __forceinline__ void tdm_load_2d(unsigned lds_addr, const void* gptr,
                                            unsigned tile_d0, unsigned tile_d1,
                                            unsigned long long stride_d0,
                                            unsigned tensor_d0, unsigned tensor_d1) {
  unsigned long long ga = (unsigned long long)(uintptr_t)gptr;
  uint32x4 g0;
  g0.x = 1u;                                               // count=1 (valid D#)
  g0.y = lds_addr;                                         // lds_addr[31:0]
  g0.z = (unsigned)(ga & 0xFFFFFFFFu);                     // global_addr[31:0]
  g0.w = (unsigned)((ga >> 32) & 0x01FFFFFFu) | (2u << 30);// addr[56:32] | type=2
  int32x8 g1;
  g1[0] = (int)(1u << 16);                                 // data_size=1 -> 2B
  g1[1] = (int)((tensor_d0 & 0xFFFFu) << 16);              // tensor_dim0[15:0]
  g1[2] = (int)((tensor_d0 >> 16) | ((tensor_d1 & 0xFFFFu) << 16));
  g1[3] = (int)((tensor_d1 >> 16) | (tile_d0 << 16));      // tile_dim0
  g1[4] = (int)(tile_d1 & 0xFFFFu);                        // tile_dim1 (dim2=0)
  g1[5] = (int)(stride_d0 & 0xFFFFFFFFull);                // dim0_stride[31:0]
  g1[6] = (int)((stride_d0 >> 32) & 0xFFFFull);            // dim0_stride[47:32]
  g1[7] = 0;
  int32x4 z4 = {0, 0, 0, 0};
#if TDM_6ARG
  int32x8 z8 = {0, 0, 0, 0, 0, 0, 0, 0};
  __builtin_amdgcn_tensor_load_to_lds(g0, g1, z4, z4, z8, 0);
#else
  __builtin_amdgcn_tensor_load_to_lds(g0, g1, z4, z4, 0);
#endif
}
#endif  // USE_TDM

// --------------------------- f32 -> bf16 (plain) ----------------------------
__global__ void cvt_bf16_kernel(const float* __restrict__ in,
                                __bf16* __restrict__ out, int n) {
  int i = blockIdx.x * blockDim.x + threadIdx.x;
  if (i < n) out[i] = (__bf16)in[i];
}

// ------------------- f32 (K x N) -> bf16 transposed (N x K) -----------------
__global__ __launch_bounds__(256) void cvt_w_t_kernel(const float* __restrict__ in,
                                                      __bf16* __restrict__ out) {
  __shared__ float tile[32][33];
  const int x  = threadIdx.x & 31;
  const int y  = threadIdx.x >> 5;  // 0..7
  const int n0 = blockIdx.x * 32;
  const int k0 = blockIdx.y * 32;
#pragma unroll
  for (int r = 0; r < 4; ++r)
    tile[y + 8 * r][x] = in[(size_t)(k0 + y + 8 * r) * DMODEL + n0 + x];
  __syncthreads();
#pragma unroll
  for (int r = 0; r < 4; ++r)
    out[(size_t)(n0 + y + 8 * r) * DMODEL + k0 + x] = (__bf16)tile[x][y + 8 * r];
}

// --------------------------- bf16 GEMM 4096x1024x1024 -----------------------
// One k-step (K=32) of the 128x64 block tile; wave computes 32x32 (2x2 WMMA).
__device__ __forceinline__ void gemm_step(const __bf16* __restrict__ As,
                                          const __bf16* __restrict__ Bst,
                                          int wm, int wn, int lq, int lh,
                                          floatx8 acc[2][2]) {
  Frag af[2];
#pragma unroll
  for (int mi = 0; mi < 2; ++mi) {
    int row = wm * 32 + mi * 16 + lq;
#pragma unroll
    for (int i = 0; i < 8; ++i) {
      int kk = (i < 4 ? 2 * i : 16 + 2 * (i - 4)) + 8 * lh;
      af[mi].u[i] = *(const unsigned int*)(As + row * 32 + kk);
    }
  }
  Frag bfr[2];
#pragma unroll
  for (int ni = 0; ni < 2; ++ni) {
    int col = wn * 32 + ni * 16 + lq;
    const uint4* p = (const uint4*)(Bst + col * 32 + 8 * lh);
    bfr[ni].q[0] = p[0];
    bfr[ni].q[1] = p[2];
  }
#pragma unroll
  for (int mi = 0; mi < 2; ++mi)
#pragma unroll
    for (int ni = 0; ni < 2; ++ni)
      acc[mi][ni] = wmma_bf16(af[mi].v, bfr[ni].v, acc[mi][ni]);
}

// A: (M x K) bf16 row-major.  Bt: (N x K) bf16 (transposed weights).
// MODE 0: out f32 row-major (M x N)
// MODE 1: out bf16 (B,H,T,hd)   from (b*T+t, h*hd+d)   [Q, K proj]
// MODE 2: out bf16 (B,H,hd,T)   from (b*T+t, h*hd+d)   [V proj, pre-transposed]
template <int MODE>
__global__ __launch_bounds__(256) void gemm_bf16_kernel(
    const __bf16* __restrict__ A, const __bf16* __restrict__ Bt,
    float* __restrict__ outf, __bf16* __restrict__ outb) {
  __shared__ __bf16 As[2][128 * 32];   // 16 KB (double-buffered M x K tile)
  __shared__ __bf16 Bst[2][64 * 32];   //  8 KB (double-buffered N x K tile)

  const int n0   = blockIdx.x * 64;
  const int m0   = blockIdx.y * 128;
  const int lane = threadIdx.x & 31;
  const int wave = threadIdx.x >> 5;
  const int lq   = lane & 15;
  const int lh   = lane >> 4;
  const int wm   = wave >> 1;
  const int wn   = wave & 1;
  const int NSTEP = DMODEL / 32;

  floatx8 acc[2][2] = {};

#if USE_TDM
  unsigned as_lds[2]  = {(unsigned)(size_t)(void*)As[0],
                         (unsigned)(size_t)(void*)As[1]};
  unsigned bst_lds[2] = {(unsigned)(size_t)(void*)Bst[0],
                         (unsigned)(size_t)(void*)Bst[1]};
  if (wave == 0) {
    tdm_load_2d(as_lds[0], A + (size_t)m0 * DMODEL, 32, 128, DMODEL, DMODEL, MROWS);
    tdm_load_2d(bst_lds[0], Bt + (size_t)n0 * DMODEL, 32, 64, DMODEL, DMODEL, DMODEL);
    __builtin_amdgcn_s_wait_tensorcnt(0);
  }
  __syncthreads();
  for (int s = 0; s < NSTEP; ++s) {
    const int cur = s & 1;
    if (wave == 0 && s + 1 < NSTEP) {
      const int kn = (s + 1) * 32;
      tdm_load_2d(as_lds[cur ^ 1], A + (size_t)m0 * DMODEL + kn, 32, 128,
                  DMODEL, DMODEL, MROWS);
      tdm_load_2d(bst_lds[cur ^ 1], Bt + (size_t)n0 * DMODEL + kn, 32, 64,
                  DMODEL, DMODEL, DMODEL);
    }
    gemm_step(As[cur], Bst[cur], wm, wn, lq, lh, acc);  // overlaps with TDM
    __syncthreads();
    if (wave == 0 && s + 1 < NSTEP) __builtin_amdgcn_s_wait_tensorcnt(0);
    __syncthreads();
  }
#else
  for (int s = 0; s < NSTEP; ++s) {
    const int k0 = s * 32;
    __syncthreads();
    {
      int c0 = threadIdx.x;
      int r0 = c0 >> 2, o0 = (c0 & 3) * 8;
      ((uint4*)As[0])[c0] = *(const uint4*)(A + (size_t)(m0 + r0) * DMODEL + k0 + o0);
      int c1 = c0 + 256;
      int r1 = c1 >> 2, o1 = (c1 & 3) * 8;
      ((uint4*)As[0])[c1] = *(const uint4*)(A + (size_t)(m0 + r1) * DMODEL + k0 + o1);
      int nb = c0 >> 2, ob = (c0 & 3) * 8;
      ((uint4*)Bst[0])[c0] = *(const uint4*)(Bt + (size_t)(n0 + nb) * DMODEL + k0 + ob);
    }
    __syncthreads();
    gemm_step(As[0], Bst[0], wm, wn, lq, lh, acc);
  }
#endif

  // C fragment: row = r + 8*half, col = lane&15
#pragma unroll
  for (int mi = 0; mi < 2; ++mi) {
#pragma unroll
    for (int ni = 0; ni < 2; ++ni) {
      int col = n0 + wn * 32 + ni * 16 + lq;
#pragma unroll
      for (int r = 0; r < 8; ++r) {
        int row = m0 + wm * 32 + mi * 16 + 8 * lh + r;
        float val = acc[mi][ni][r];
        if (MODE == 0) {
          outf[(size_t)row * DMODEL + col] = val;
        } else {
          int bb = row >> 11;            // / T_SEQ
          int tp = row & (T_SEQ - 1);
          int hd = col >> 6;             // / HDIM
          int dd = col & (HDIM - 1);
          if (MODE == 1)
            outb[(((size_t)(bb * NHEAD + hd) * T_SEQ) + tp) * HDIM + dd] = (__bf16)val;
          else
            outb[(((size_t)(bb * NHEAD + hd) * HDIM) + dd) * T_SEQ + tp] = (__bf16)val;
        }
      }
    }
  }
}

// --------------------------- fused flash attention --------------------------
// One 32-key step for a wave's 16-query tile. Q-fragments are pre-scaled by
// 1/sqrt(hd), so scores need only the additive ALiBi bias.
__device__ __forceinline__ void flash_step(
    const __bf16* __restrict__ Ksb, const __bf16* __restrict__ Vsb,
    const Frag bq[2], floatx8 o[4], float& m, float& l,
    int k0, int qg, int lh, int lq, float slope, const float* srf) {
  // S^T tiles (keys x queries): load all 4 A-fragments, then 4 WMMAs
  Frag ka[2][2];
#pragma unroll
  for (int kt = 0; kt < 2; ++kt) {
    int key = kt * 16 + lq;
#pragma unroll
    for (int c = 0; c < 2; ++c)
#pragma unroll
      for (int i = 0; i < 8; ++i) {
        int kk = c * 32 + (i < 4 ? 2 * i : 16 + 2 * (i - 4)) + 8 * lh;
        ka[kt][c].u[i] = *(const unsigned int*)(Ksb + key * 64 + kk);
      }
  }
  floatx8 sc[2] = {};
#pragma unroll
  for (int kt = 0; kt < 2; ++kt)
#pragma unroll
    for (int c = 0; c < 2; ++c)
      sc[kt] = wmma_bf16(ka[kt][c].v, bq[c].v, sc[kt]);

  // ALiBi + causal mask; lane holds keys (k0 + kt*16 + 8*half + r) for query qg
  float biasb[2];
  int   thr[2];
#pragma unroll
  for (int kt = 0; kt < 2; ++kt) {
    int keybase = k0 + kt * 16 + 8 * lh;
    biasb[kt] = slope * (float)(keybase - (T_SEQ - 1));
    thr[kt]   = qg - keybase;  // keep key r iff r <= thr
  }
  float pv[2][8];
  float locmax = -1e30f;
#pragma unroll
  for (int kt = 0; kt < 2; ++kt)
#pragma unroll
    for (int r = 0; r < 8; ++r) {
      float sv = sc[kt][r] + biasb[kt] + srf[r];
      sv = (r <= thr[kt]) ? sv : -1e30f;
      pv[kt][r] = sv;
      locmax = fmaxf(locmax, sv);
    }
  float om    = __shfl_xor(locmax, 16, 32);
  float mnew  = fmaxf(m, fmaxf(locmax, om));
  float alpha = __expf(m - mnew);
  float lsum  = 0.f;
#pragma unroll
  for (int kt = 0; kt < 2; ++kt)
#pragma unroll
    for (int r = 0; r < 8; ++r) {
      float p = __expf(pv[kt][r] - mnew);  // masked: exp(-1e30-m) underflows to 0
      pv[kt][r] = p;
      lsum += p;
    }
  lsum += __shfl_xor(lsum, 16, 32);
  l = l * alpha + lsum;
  m = mnew;

  // P^T B-fragment (32 keys x 16 q) from the S^T C-fragments, one exchange
  float p0x[8], p1x[8];
#pragma unroll
  for (int r = 0; r < 8; ++r) {
    p0x[r] = __shfl_xor(pv[0][r], 16, 32);
    p1x[r] = __shfl_xor(pv[1][r], 16, 32);
  }
  Frag pb;
#pragma unroll
  for (int j = 0; j < 8; ++j)
    pb.h[j] = (__bf16)(lh ? p1x[j] : pv[0][j]);
#pragma unroll
  for (int j = 0; j < 8; ++j)
    pb.h[8 + j] = (__bf16)(lh ? pv[1][j] : p0x[j]);

  // O^T += V^T(16hd x 32keys) @ P^T(32keys x 16q): load 4 frags, then 4 WMMAs
  Frag va[4];
#pragma unroll
  for (int t = 0; t < 4; ++t) {
    int hd = t * 16 + lq;
#pragma unroll
    for (int i = 0; i < 8; ++i) {
      int kk = 2 * i + 16 * lh;
      va[t].u[i] = *(const unsigned int*)(Vsb + hd * 32 + kk);
    }
  }
#pragma unroll
  for (int t = 0; t < 4; ++t) {
    o[t] = o[t] * alpha;
    o[t] = wmma_bf16(va[t].v, pb.v, o[t]);
  }
}

// grid = (T/128, B*H), block = 256 (8 waves). Wave w handles q rows [q0,q0+16).
// Q/K layout (B,H,T,hd); V layout (B,H,hd,T) (pre-transposed by projection).
__global__ __launch_bounds__(256) void flash_alibi_kernel(
    const __bf16* __restrict__ Qg, const __bf16* __restrict__ Kg,
    const __bf16* __restrict__ Vtg, __bf16* __restrict__ attnb) {
  __shared__ __bf16 Ks[2][32 * 64];   // keys x hd (double-buffered)
  __shared__ __bf16 Vst[2][64 * 32];  // hd x keys (double-buffered)

  const int bh = blockIdx.y;
  const int bb = bh >> 4;   // batch
  const int hh = bh & 15;   // head
  const __bf16* Qp  = Qg  + (size_t)bh * T_SEQ * HDIM;
  const __bf16* Kp  = Kg  + (size_t)bh * T_SEQ * HDIM;
  const __bf16* Vtp = Vtg + (size_t)bh * HDIM * T_SEQ;

  const int lane = threadIdx.x & 31;
  const int wave = threadIdx.x >> 5;
  const int lq   = lane & 15;
  const int lh   = lane >> 4;
  const int q0   = blockIdx.x * 128 + wave * 16;
  const int qg   = q0 + lq;
  // wave-uniform causal bound as an SGPR -> scalar branch, EXEC stays all-ones
  const int qtop = (int)__builtin_amdgcn_readfirstlane(q0 + 15);
  // H=16 (power of 2): slopes[h] = 2^(-0.5*(h+1))
  const float slope = exp2f(-0.5f * (float)(hh + 1));
  float srf[8];
#pragma unroll
  for (int r = 0; r < 8; ++r) srf[r] = slope * (float)r;

  // Q^T B-fragments, pre-scaled by 1/sqrt(hd)=0.125 (exact: power of two)
  Frag bq[2];
#pragma unroll
  for (int c = 0; c < 2; ++c) {
    const uint4* p = (const uint4*)(Qp + (size_t)qg * HDIM + c * 32 + lh * 8);
    bq[c].q[0] = p[0];
    bq[c].q[1] = p[2];
#pragma unroll
    for (int e = 0; e < 16; ++e)
      bq[c].h[e] = (__bf16)((float)bq[c].h[e] * 0.125f);
  }

  floatx8 o[4] = {};           // O^T tiles: hd rows 16t + r + 8*half, col = q
  float m = -1e30f, l = 0.f;   // online softmax state (per query column)

  const int nsteps = (blockIdx.x * 128 + 127) / 32 + 1;

#if USE_TDM
  unsigned ks_lds[2]  = {(unsigned)(size_t)(void*)Ks[0],
                         (unsigned)(size_t)(void*)Ks[1]};
  unsigned vst_lds[2] = {(unsigned)(size_t)(void*)Vst[0],
                         (unsigned)(size_t)(void*)Vst[1]};
  if (wave == 0) {
    tdm_load_2d(ks_lds[0], Kp, HDIM, 32, HDIM, HDIM, T_SEQ);
    tdm_load_2d(vst_lds[0], Vtp, 32, HDIM, T_SEQ, T_SEQ, HDIM);
    __builtin_amdgcn_s_wait_tensorcnt(0);
  }
  __syncthreads();
  for (int s = 0; s < nsteps; ++s) {
    const int k0  = s * 32;
    const int cur = s & 1;
    if (wave == 0 && s + 1 < nsteps) {
      const int kn = (s + 1) * 32;
      tdm_load_2d(ks_lds[cur ^ 1], Kp + (size_t)kn * HDIM, HDIM, 32,
                  HDIM, HDIM, T_SEQ);
      tdm_load_2d(vst_lds[cur ^ 1], Vtp + kn, 32, HDIM, T_SEQ, T_SEQ, HDIM);
    }
    if (k0 <= qtop)  // scalar compare: skip fully-masked key blocks
      flash_step(Ks[cur], Vst[cur], bq, o, m, l, k0, qg, lh, lq, slope, srf);
    __syncthreads();
    if (wave == 0 && s + 1 < nsteps) __builtin_amdgcn_s_wait_tensorcnt(0);
    __syncthreads();
  }
#else
  for (int s = 0; s < nsteps; ++s) {
    const int k0 = s * 32;
    __syncthreads();
    {
      int trow = threadIdx.x >> 3;
      int tco  = (threadIdx.x & 7) * 8;
      *(uint4*)(Ks[0] + trow * 64 + tco) =
          *(const uint4*)(Kp + (size_t)(k0 + trow) * HDIM + tco);
      int hrow = threadIdx.x >> 2;
      int kco  = (threadIdx.x & 3) * 8;
      *(uint4*)(Vst[0] + hrow * 32 + kco) =
          *(const uint4*)(Vtp + (size_t)hrow * T_SEQ + k0 + kco);
    }
    __syncthreads();
    if (k0 <= qtop)
      flash_step(Ks[0], Vst[0], bq, o, m, l, k0, qg, lh, lq, slope, srf);
  }
#endif

  // normalize + store to attn (B,T,D) layout, column = h*64 + hd
  float inv = 1.0f / l;
#pragma unroll
  for (int t = 0; t < 4; ++t) {
    V8 st;
#pragma unroll
    for (int r = 0; r < 8; ++r) st.h[r] = (__bf16)(o[t][r] * inv);
    *(uint4*)(attnb + ((size_t)bb * T_SEQ + qg) * DMODEL +
              hh * HDIM + t * 16 + 8 * lh) = st.q;
  }
}

// ------------------------------- launcher -----------------------------------
extern "C" void kernel_launch(void* const* d_in, const int* in_sizes, int n_in,
                              void* d_out, int out_size, void* d_ws, size_t ws_size,
                              hipStream_t stream) {
  (void)in_sizes; (void)n_in; (void)out_size; (void)ws_size;
  const float* x  = (const float*)d_in[0];
  const float* Wq = (const float*)d_in[1];
  const float* Wk = (const float*)d_in[2];
  const float* Wv = (const float*)d_in[3];
  const float* Wo = (const float*)d_in[4];

  __bf16* ws = (__bf16*)d_ws;
  const size_t NX = (size_t)MROWS * DMODEL;    // 4,194,304
  const size_t NW = (size_t)DMODEL * DMODEL;   // 1,048,576
  __bf16* xb  = ws;                 // x bf16
  __bf16* wqt = xb + NX;            // transposed bf16 weights (N x K)
  __bf16* wkt = wqt + NW;
  __bf16* wvt = wkt + NW;
  __bf16* wot = wvt + NW;
  __bf16* qb  = wot + NW;           // Q (B,H,T,hd)
  __bf16* kb  = qb + NX;            // K (B,H,T,hd)
  __bf16* vtb = kb + NX;            // V (B,H,hd,T)  pre-transposed
  __bf16* ab  = vtb + NX;           // attn_out (B,T,D)
  // total: 25,165,824 bf16 = 48 MB of workspace

  cvt_bf16_kernel<<<(int)((NX + 255) / 256), 256, 0, stream>>>(x, xb, (int)NX);
  dim3 gt(DMODEL / 32, DMODEL / 32);
  cvt_w_t_kernel<<<gt, 256, 0, stream>>>(Wq, wqt);
  cvt_w_t_kernel<<<gt, 256, 0, stream>>>(Wk, wkt);
  cvt_w_t_kernel<<<gt, 256, 0, stream>>>(Wv, wvt);
  cvt_w_t_kernel<<<gt, 256, 0, stream>>>(Wo, wot);

  dim3 gg(DMODEL / 64, MROWS / 128);
  gemm_bf16_kernel<1><<<gg, 256, 0, stream>>>(xb, wqt, nullptr, qb);
  gemm_bf16_kernel<1><<<gg, 256, 0, stream>>>(xb, wkt, nullptr, kb);
  gemm_bf16_kernel<2><<<gg, 256, 0, stream>>>(xb, wvt, nullptr, vtb);

  flash_alibi_kernel<<<dim3(T_SEQ / 128, 2 * NHEAD), 256, 0, stream>>>(qb, kb, vtb, ab);

  gemm_bf16_kernel<0><<<gg, 256, 0, stream>>>(ab, wot, (float*)d_out, nullptr);
}
